// NeRF_73323681677984
// MI455X (gfx1250) — matrix operand order repaired and checked by reading, not compiled, not run on hardware
//
#include <hip/hip_runtime.h>
#include <hip/hip_bf16.h>

// ---------------- types ----------------
typedef __attribute__((ext_vector_type(16))) _Float16 v16h;
typedef __attribute__((ext_vector_type(8)))  _Float16 v8h;
typedef __attribute__((ext_vector_type(8)))  float    v8f;

#define NTOT   524288
#define NSAMP  128          // samples per workgroup
#define AST    264          // activation LDS stride (halfs), padded (132 dwords -> conflict-free)
#define EPST   64           // enc_pos LDS stride (63 -> 64)
#define EDST   32           // enc_dir LDS stride (27 -> 32)

// fp16 weight blob offsets in d_ws (halfs); every K padded to /32, rows padded to /16
constexpr size_t OW0  = 0;                    // [256][64]   (63 -> 64)
constexpr size_t OW1  = OW0  + 256 * 64;      // [256][256]
constexpr size_t OW2  = OW1  + 256 * 256;
constexpr size_t OW3  = OW2  + 256 * 256;
constexpr size_t OW4  = OW3  + 256 * 256;     // [256][320]  (319 -> 320, skip layer)
constexpr size_t OW5  = OW4  + 256 * 320;
constexpr size_t OW6  = OW5  + 256 * 256;
constexpr size_t OW7  = OW6  + 256 * 256;
constexpr size_t OWf  = OW7  + 256 * 256;     // [256][256]
constexpr size_t OWs  = OWf  + 256 * 256;     // [16][256]   (1 -> 16 rows)
constexpr size_t OWc1 = OWs  + 16  * 256;     // [128][288]  (283 -> 288)
constexpr size_t OWc2 = OWc1 + 128 * 288;     // [16][128]   (3 -> 16 rows)
constexpr size_t BLOB_HALFS = OWc2 + 16 * 128;

constexpr unsigned SMEM_BYTES =
    (2u * NSAMP * AST + NSAMP * EPST + NSAMP * EDST) * sizeof(_Float16); // 159744 B

// ---------------- fragment helpers ----------------
union V16U { v16h v; v8h h[2]; };

// B (32x16): lane lh=N, ls selects K 0-15 / 16-31; 16 contiguous halfs of a weight row
__device__ __forceinline__ v16h load_b_frag(const _Float16* p) {
  V16U u; u.h[0] = *(const v8h*)p; u.h[1] = *(const v8h*)(p + 8); return u.v;
}
// A (16x32): lane lh=M, ls=0 -> K{0..15}, ls=1 -> K{16..31}; p already has +ls*16
__device__ __forceinline__ v16h load_a_frag(const _Float16* p) {
  V16U u; u.h[0] = *(const v8h*)p; u.h[1] = *(const v8h*)(p + 8); return u.v;
}
__device__ __forceinline__ v8f wmma_f16(v16h a, v16h b, v8f c) {
  return __builtin_amdgcn_wmma_f32_16x16x32_f16(false, a, false, b, (short)0, c, false, false);
}

// ---------------- generic fused dense layer ----------------
// Wave owns channel-tiles [cg*TCT, cg*TCT+TCT) x sample-tiles [sg*TST, sg*TST+TST).
// A-fragments (LDS) are reused across TCT WMMAs; B-fragments (global/L2 weights)
// are reused across TST WMMAs and live in registers.
// K is split into 32-wide chunks: c0 chunks from a0 (stride a0st), then c1 from a1.
template<int TCT, int TST, bool RELU>
__device__ __forceinline__ void dense_layer(
    const _Float16* __restrict__ Wg, int kpad,
    const float* __restrict__ bias,
    const _Float16* __restrict__ a0, int a0st, int c0,
    const _Float16* __restrict__ a1, int a1st, int c1,
    _Float16* __restrict__ ho, int hst,
    int cg, int sg, int lane)
{
  const int lh = lane & 15;   // N / channel within tile
  const int ls = lane >> 4;   // K-half select

  v8f acc[TST][TCT];
#pragma unroll
  for (int j = 0; j < TCT; ++j) {
    const float bv = bias[(cg * TCT + j) * 16 + lh];
#pragma unroll
    for (int t = 0; t < TST; ++t)
#pragma unroll
      for (int r = 0; r < 8; ++r)
        acc[t][j][r] = bv;
  }

  const int total = c0 + c1;
  for (int kc = 0; kc < total; ++kc) {
    const _Float16* asrc; int astr; int kl;
    if (kc < c0) { asrc = a0; astr = a0st; kl = kc; }
    else         { asrc = a1; astr = a1st; kl = kc - c0; }

    v16h bfr[TCT];
#pragma unroll
    for (int j = 0; j < TCT; ++j) {
      const _Float16* wp = Wg + (size_t)((cg * TCT + j) * 16 + lh) * kpad
                              + kc * 32 + ls * 16;
      if (kc + 1 < total) __builtin_prefetch(wp + 32, 0, 1);  // next K-chunk (L2 hot)
      bfr[j] = load_b_frag(wp);
    }
#pragma unroll
    for (int t = 0; t < TST; ++t) {
      const int st = sg * TST + t;
      v16h afr = load_a_frag(asrc + (st * 16 + lh) * astr + kl * 32 + ls * 16);
#pragma unroll
      for (int j = 0; j < TCT; ++j)
        acc[t][j] = wmma_f16(afr, bfr[j], acc[t][j]);
    }
  }

  // C/D layout: VGPR r -> row M = r + ls*8, col N = lh
#pragma unroll
  for (int t = 0; t < TST; ++t)
#pragma unroll
    for (int j = 0; j < TCT; ++j) {
      const int ch  = (cg * TCT + j) * 16 + lh;
      const int row = (sg * TST + t) * 16 + ls * 8;
#pragma unroll
      for (int r = 0; r < 8; ++r) {
        float v = acc[t][j][r];
        if (RELU) v = fmaxf(v, 0.0f);
        ho[(row + r) * hst + ch] = (_Float16)v;
      }
    }
}

// single 16x16 output tile head (wave handles sample-tile `st`)
__device__ __forceinline__ v8f head_tile(
    const _Float16* __restrict__ Wg, int kpad,
    const _Float16* __restrict__ a, int astr, int chunks,
    float bias_lane, int st, int lane)
{
  const int lh = lane & 15, ls = lane >> 4;
  v8f acc;
#pragma unroll
  for (int r = 0; r < 8; ++r) acc[r] = bias_lane;
  for (int kc = 0; kc < chunks; ++kc) {
    v16h bfr = load_b_frag(Wg + (size_t)lh * kpad + kc * 32 + ls * 16);
    v16h afr = load_a_frag(a + (st * 16 + lh) * astr + kc * 32 + ls * 16);
    acc = wmma_f16(afr, bfr, acc);
  }
  return acc;
}

// ---------------- fp32 -> fp16 weight pad/convert ----------------
__global__ void cvt_pad(const float* __restrict__ src, _Float16* __restrict__ dst,
                        int osrc, int isrc, int idst, int total)
{
  int i = blockIdx.x * 256 + threadIdx.x;
  if (i >= total) return;
  int o = i / idst, k = i - o * idst;
  float v = (o < osrc && k < isrc) ? src[(size_t)o * isrc + k] : 0.0f;
  dst[i] = (_Float16)v;
}

// ---------------- fully fused NeRF kernel ----------------
__global__ __launch_bounds__(256) void nerf_fused(
    const float* __restrict__ ep, const float* __restrict__ ed,
    const _Float16* __restrict__ blob,
    const float* __restrict__ b0, const float* __restrict__ b1,
    const float* __restrict__ b2, const float* __restrict__ b3,
    const float* __restrict__ b4, const float* __restrict__ b5,
    const float* __restrict__ b6, const float* __restrict__ b7,
    const float* __restrict__ biasf, const float* __restrict__ biass,
    const float* __restrict__ biasc1, const float* __restrict__ biasc2,
    float* __restrict__ out)
{
  extern __shared__ _Float16 smem[];
  _Float16* buf0 = smem;                       // [128][AST]
  _Float16* buf1 = buf0 + NSAMP * AST;         // [128][AST]
  _Float16* sep  = buf1 + NSAMP * AST;         // [128][64]  enc_pos (persistent)
  _Float16* sed  = sep  + NSAMP * EPST;        // [128][32]  enc_dir (persistent)

  const int tid   = threadIdx.x;
  const int lane  = tid & 31;
  const int wave  = tid >> 5;
  const int gbase = blockIdx.x * NSAMP;

  // trunk tiling: 4 channel-groups x 2 sample-groups (TCT=4, TST=4)
  const int cg4 = wave & 3, sg4 = wave >> 2;
  // Wc1 tiling: 2 channel-groups x 4 sample-groups (TCT=4, TST=2)
  const int cg2 = wave & 1, sg2 = wave >> 1;

  // stage encodings -> fp16 LDS (zero-padded)
  for (int i = tid; i < NSAMP * EPST; i += 256) {
    int s = i >> 6, c = i & 63;
    float v = (c < 63) ? ep[(size_t)(gbase + s) * 63 + c] : 0.0f;
    sep[i] = (_Float16)v;
  }
  for (int i = tid; i < NSAMP * EDST; i += 256) {
    int s = i >> 5, c = i & 31;
    float v = (c < 27) ? ed[(size_t)(gbase + s) * 27 + c] : 0.0f;
    sed[i] = (_Float16)v;
  }
  __syncthreads();

  // trunk MLP (ping-pong buffers)
  dense_layer<4, 4, true >(blob + OW0,  64, b0, sep,  EPST, 2, nullptr, 0, 0, buf0, AST, cg4, sg4, lane);
  __syncthreads();
  dense_layer<4, 4, true >(blob + OW1, 256, b1, buf0, AST,  8, nullptr, 0, 0, buf1, AST, cg4, sg4, lane);
  __syncthreads();
  dense_layer<4, 4, true >(blob + OW2, 256, b2, buf1, AST,  8, nullptr, 0, 0, buf0, AST, cg4, sg4, lane);
  __syncthreads();
  dense_layer<4, 4, true >(blob + OW3, 256, b3, buf0, AST,  8, nullptr, 0, 0, buf1, AST, cg4, sg4, lane);
  __syncthreads();
  // skip layer: concat(h, enc_pos) -> K = 256 + 64
  dense_layer<4, 4, true >(blob + OW4, 320, b4, buf1, AST,  8, sep, EPST, 2, buf0, AST, cg4, sg4, lane);
  __syncthreads();
  dense_layer<4, 4, true >(blob + OW5, 256, b5, buf0, AST,  8, nullptr, 0, 0, buf1, AST, cg4, sg4, lane);
  __syncthreads();
  dense_layer<4, 4, true >(blob + OW6, 256, b6, buf1, AST,  8, nullptr, 0, 0, buf0, AST, cg4, sg4, lane);
  __syncthreads();
  dense_layer<4, 4, true >(blob + OW7, 256, b7, buf0, AST,  8, nullptr, 0, 0, buf1, AST, cg4, sg4, lane);
  __syncthreads();

  // feature = h @ Wf^T + bf  (no relu), h in buf1 -> buf0
  dense_layer<4, 4, false>(blob + OWf, 256, biasf, buf1, AST, 8, nullptr, 0, 0, buf0, AST, cg4, sg4, lane);

  // sigma = h @ Ws^T + bs : one 16x16 tile per wave (only column 0 valid)
  {
    const int lh = lane & 15, ls = lane >> 4;
    const float bl = (lh == 0) ? biass[0] : 0.0f;
    v8f acc = head_tile(blob + OWs, 256, buf1, AST, 8, bl, wave, lane);
    if (lh == 0) {
#pragma unroll
      for (int r = 0; r < 8; ++r)
        out[(size_t)(gbase + wave * 16 + ls * 8 + r) * 4 + 3] = acc[r];
    }
  }
  __syncthreads();

  // color hidden = relu(concat(feature, enc_dir) @ Wc1^T + bc1), 128 out -> buf1
  dense_layer<4, 2, true >(blob + OWc1, 288, biasc1, buf0, AST, 8, sed, EDST, 1, buf1, AST, cg2, sg2, lane);
  __syncthreads();

  // color = hidden @ Wc2^T + bc2 : one tile per wave (columns 0..2 valid)
  {
    const int lh = lane & 15, ls = lane >> 4;
    const float bl = (lh < 3) ? biasc2[lh] : 0.0f;
    v8f acc = head_tile(blob + OWc2, 128, buf1, AST, 4, bl, wave, lane);
    if (lh < 3) {
#pragma unroll
      for (int r = 0; r < 8; ++r)
        out[(size_t)(gbase + wave * 16 + ls * 8 + r) * 4 + lh] = acc[r];
    }
  }
}

// ---------------- launcher ----------------
extern "C" void kernel_launch(void* const* d_in, const int* in_sizes, int n_in,
                              void* d_out, int out_size, void* d_ws, size_t ws_size,
                              hipStream_t stream)
{
  (void)in_sizes; (void)n_in; (void)out_size; (void)ws_size;

  const float* ep = (const float*)d_in[0];
  const float* ed = (const float*)d_in[1];
  _Float16* blob  = (_Float16*)d_ws;

  struct Spec { int idx, osrc, isrc, odst, idst; size_t off; };
  const Spec specs[] = {
    { 2, 256,  63, 256,  64, OW0  },
    { 4, 256, 256, 256, 256, OW1  },
    { 6, 256, 256, 256, 256, OW2  },
    { 8, 256, 256, 256, 256, OW3  },
    {10, 256, 319, 256, 320, OW4  },
    {12, 256, 256, 256, 256, OW5  },
    {14, 256, 256, 256, 256, OW6  },
    {16, 256, 256, 256, 256, OW7  },
    {18, 256, 256, 256, 256, OWf  },
    {20,   1, 256,  16, 256, OWs  },
    {22, 128, 283, 128, 288, OWc1 },
    {24,   3, 128,  16, 128, OWc2 },
  };
  for (const Spec& s : specs) {
    int total = s.odst * s.idst;
    cvt_pad<<<(total + 255) / 256, 256, 0, stream>>>(
        (const float*)d_in[s.idx], blob + s.off, s.osrc, s.isrc, s.idst, total);
  }

  (void)hipFuncSetAttribute((const void*)nerf_fused,
                            hipFuncAttributeMaxDynamicSharedMemorySize,
                            (int)SMEM_BYTES);

  nerf_fused<<<NTOT / NSAMP, 256, SMEM_BYTES, stream>>>(
      ep, ed, blob,
      (const float*)d_in[3],  (const float*)d_in[5],
      (const float*)d_in[7],  (const float*)d_in[9],
      (const float*)d_in[11], (const float*)d_in[13],
      (const float*)d_in[15], (const float*)d_in[17],
      (const float*)d_in[19], (const float*)d_in[21],
      (const float*)d_in[23], (const float*)d_in[25],
      (float*)d_out);
}
